// DeformableTransformerDecoderLayer_28458453303457
// MI455X (gfx1250) — compile-verified, matrix operand order checked by reading, not run on hardware
//
#include <hip/hip_runtime.h>
#include <hip/hip_bf16.h>

// ---------------------------------------------------------------------------
// SeqFormer DeformableTransformerDecoderLayer for MI455X (gfx1250, wave32).
// GEMMs: activations/weights pre-converted to bf16 (RNE) by a streaming pass,
// then v_wmma_f32_16x16x32_bf16 with one 16x128 output strip per wave.
// K is a template constant so all 8 B-fragment addresses fold into immediate
// offsets (one s_clause of b128 loads), and all B fragments are fetched before
// the 8 WMMAs so loadcnt waits stagger instead of draining per-WMMA.
// ---------------------------------------------------------------------------

#define D_MODEL 256
#define D_FFN   1024
#define NF      5
#define LQ      300
#define NHEAD   8
#define HDIM    32
#define NLVL    4
#define NPTS    4
#define LEN_IN  20197

__constant__ int cH[4] = {100, 50, 25, 13};
__constant__ int cW[4] = {152, 76, 38, 19};
__constant__ int cS[4] = {0, 15200, 19000, 19950};

typedef __attribute__((ext_vector_type(16))) __bf16       bf16x16;
typedef __attribute__((ext_vector_type(8)))  float        f32x8;
typedef __attribute__((ext_vector_type(4)))  unsigned int u32x4;
typedef __attribute__((ext_vector_type(8)))  unsigned int u32x8;
typedef unsigned short bf16_t;   // bf16 storage

// Pack high halves of two f32 bit patterns into one bf16 pair (one v_perm_b32)
static __device__ __forceinline__ unsigned int pkbf(unsigned int hi, unsigned int lo) {
  return __builtin_amdgcn_perm(hi, lo, 0x07060302u);
}

// ---------------------------------------------------------------------------
// Streaming f32 -> bf16 (round-to-nearest-even), 8 elements/thread.
// n must be a multiple of 8 (true at every call site).
// ---------------------------------------------------------------------------
__global__ void cvt_bf16(const float* __restrict__ in, bf16_t* __restrict__ out, int n) {
  const int i = (blockIdx.x * blockDim.x + threadIdx.x) * 8;
  if (i >= n) return;
  u32x4 a = *(const u32x4*)(in + i);
  u32x4 b = *(const u32x4*)(in + i + 4);
#pragma unroll
  for (int j = 0; j < 4; ++j) {
    a[j] = a[j] + 0x7FFFu + ((a[j] >> 16) & 1u);
    b[j] = b[j] + 0x7FFFu + ((b[j] >> 16) & 1u);
  }
  u32x4 r;
  r[0] = pkbf(a[1], a[0]); r[1] = pkbf(a[3], a[2]);
  r[2] = pkbf(b[1], b[0]); r[3] = pkbf(b[3], b[2]);
  *(u32x4*)(out + i) = r;
}

// Load a 16-element bf16 fragment: per CDNA5 16-bit 16x32 layout this lane's
// K offsets are {kb..kb+7, kb+16..kb+23} -> two contiguous b128 loads.
static __device__ __forceinline__ bf16x16 load_frag(const bf16_t* __restrict__ p) {
  u32x4 lo = *(const u32x4*)(p);        // K = kb .. kb+7
  u32x4 hi = *(const u32x4*)(p + 16);   // K = kb+16 .. kb+23
  u32x8 r;
  r[0] = lo[0]; r[1] = lo[1]; r[2] = lo[2]; r[3] = lo[3];
  r[4] = hi[0]; r[5] = hi[1]; r[6] = hi[2]; r[7] = hi[3];
  return __builtin_bit_cast(bf16x16, r);
}

// ---------------------------------------------------------------------------
// C[M,N] = A[M,K] @ W[N,K]^T + bias[N]   (optional ReLU), A/W in bf16.
// One wave computes a 16 x (16*NT) strip. Per K-step: 2 b128 loads for the
// shared A fragment, 2*NT b128 loads (immediate offsets, one clause) for the
// B fragments, then NT WMMAs on independent accumulators. Out-of-range A rows
// are clamped (their D rows are never stored) so every load is unconditional.
// Requires: N % (16*NT) == 0, K % 32 == 0 (true at every call site).
// ---------------------------------------------------------------------------
template <int NT, int K>
__global__ void gemm_bf16_wmma(const bf16_t* __restrict__ A,
                               const bf16_t* __restrict__ W,
                               const float* __restrict__ bias,
                               float* __restrict__ C,
                               int M, int N, int relu) {
  const int lane = threadIdx.x & 31;
  const int wave = threadIdx.x >> 5;
  const int nblk = N / (16 * NT);
  const int tilesM = (M + 15) >> 4;
  const int blk = blockIdx.x * (blockDim.x >> 5) + wave;
  if (blk >= tilesM * nblk) return;
  const int tm = (blk / nblk) << 4;
  const int tn = (blk % nblk) * (16 * NT);

  const int half = lane >> 4;        // which K-octet group this lane owns
  const int l16  = lane & 15;
  int arow = tm + l16;
  if (arow >= M) arow = M - 1;       // clamp: harmless, row never stored
  const bf16_t* __restrict__ ap  = A + (size_t)arow * K;
  const bf16_t* __restrict__ bp0 = W + (size_t)(tn + l16) * K;

  f32x8 acc[NT];
#pragma unroll
  for (int nt = 0; nt < NT; ++nt) acc[nt] = {};

  for (int k0 = 0; k0 < K; k0 += 32) {
    const int kb = k0 + half * 8;
    if (k0 + 32 < K) __builtin_prefetch(ap + k0 + 32, 0, 3);  // global_prefetch_b8
    const bf16x16 av = load_frag(ap + kb);
    bf16x16 bv[NT];
#pragma unroll
    for (int nt = 0; nt < NT; ++nt)              // nt*16*K is a compile-time
      bv[nt] = load_frag(bp0 + nt * 16 * K + kb); // constant -> imm offsets
#pragma unroll
    for (int nt = 0; nt < NT; ++nt)
      acc[nt] = __builtin_amdgcn_wmma_f32_16x16x32_bf16(
          /*neg_a=*/false, av, /*neg_b=*/false, bv[nt],
          /*c_mod=*/(short)0, acc[nt], /*reuse_a=*/false, /*reuse_b=*/false);
  }

  // C/D layout: VGPR r, lane<16 -> (M=tm+r, N=lane); lane>=16 -> (M=tm+r+8)
#pragma unroll
  for (int nt = 0; nt < NT; ++nt) {
    const int ocol = tn + nt * 16 + l16;
    const float bb = bias ? bias[ocol] : 0.f;
#pragma unroll
    for (int r = 0; r < 8; ++r) {
      const int orow = tm + r + half * 8;
      if (orow < M) {
        float v = acc[nt][r] + bb;
        if (relu) v = v > 0.f ? v : 0.f;
        C[(size_t)orow * N + ocol] = v;
      }
    }
  }
}

// ---------------------------------------------------------------------------
// MHA core: one wave per (batch, head, query); lane == head dim (HDIM==32).
// Online softmax over Lk keys. Q/K/V are [B*L, 256] with head h at chans h*32.
// ---------------------------------------------------------------------------
__global__ void mha_attn(const float* __restrict__ Q, const float* __restrict__ Kt,
                         const float* __restrict__ V, float* __restrict__ O,
                         int Bn, int Lq, int Lk) {
  const int gw = (int)((blockIdx.x * blockDim.x + threadIdx.x) >> 5);
  const int lane = threadIdx.x & 31;
  const int total = Bn * NHEAD * Lq;
  if (gw >= total) return;
  const int q = gw % Lq;
  const int h = (gw / Lq) % NHEAD;
  const int b = gw / (Lq * NHEAD);

  const float qd = Q[((size_t)(b * Lq + q) * D_MODEL) + h * HDIM + lane] * 0.17677669529663687f;
  float m = -3.4e38f, s = 0.f, acc = 0.f;
  for (int j = 0; j < Lk; ++j) {
    const size_t kr = (size_t)(b * Lk + j) * D_MODEL + h * HDIM;
    float t = qd * Kt[kr + lane];
    for (int off = 16; off > 0; off >>= 1) t += __shfl_xor(t, off, 32);
    const float mn = fmaxf(m, t);
    const float sc = __expf(m - mn);
    const float e  = __expf(t - mn);
    s   = s * sc + e;
    acc = acc * sc + e * V[kr + lane];
    m = mn;
  }
  O[((size_t)(b * Lq + q) * D_MODEL) + h * HDIM + lane] = acc / s;
}

// out[row] = LayerNorm(X[row] + Y[row]) * g + b ; one wave per 256-wide row
__global__ void add_ln(const float* __restrict__ X, const float* __restrict__ Y,
                       const float* __restrict__ g, const float* __restrict__ bta,
                       float* __restrict__ out, int rows) {
  const int r = (int)((blockIdx.x * blockDim.x + threadIdx.x) >> 5);
  const int lane = threadIdx.x & 31;
  if (r >= rows) return;
  const size_t base = (size_t)r * D_MODEL;
  float v[8];
  float s = 0.f;
#pragma unroll
  for (int i = 0; i < 8; ++i) {
    v[i] = X[base + lane + 32 * i] + Y[base + lane + 32 * i];
    s += v[i];
  }
  for (int off = 16; off > 0; off >>= 1) s += __shfl_xor(s, off, 32);
  const float mean = s * (1.f / 256.f);
  float ss = 0.f;
#pragma unroll
  for (int i = 0; i < 8; ++i) { const float d = v[i] - mean; ss += d * d; }
  for (int off = 16; off > 0; off >>= 1) ss += __shfl_xor(ss, off, 32);
  const float rstd = rsqrtf(ss * (1.f / 256.f) + 1e-5f);
#pragma unroll
  for (int i = 0; i < 8; ++i) {
    const int c = lane + 32 * i;
    out[base + c] = (v[i] - mean) * rstd * g[c] + bta[c];
  }
}

// o[i] = a[i] + b[i % bmod]   (broadcast add, e.g. query_pos over frames)
__global__ void add_bc(const float* __restrict__ a, const float* __restrict__ b,
                       float* __restrict__ o, int n, int bmod) {
  const int i = blockIdx.x * blockDim.x + threadIdx.x;
  if (i < n) o[i] = a[i] + b[i % bmod];
}

// softmax over groups of 16 (attention-weight softmax per head)
__global__ void softmax16(const float* __restrict__ in, float* __restrict__ out, int rows) {
  const int r = blockIdx.x * blockDim.x + threadIdx.x;
  if (r >= rows) return;
  const float* x = in + (size_t)r * 16;
  float m = -3.4e38f;
#pragma unroll
  for (int i = 0; i < 16; ++i) m = fmaxf(m, x[i]);
  float v[16], s = 0.f;
#pragma unroll
  for (int i = 0; i < 16; ++i) { v[i] = __expf(x[i] - m); s += v[i]; }
  const float inv = 1.f / s;
  float* o = out + (size_t)r * 16;
#pragma unroll
  for (int i = 0; i < 16; ++i) o[i] = v[i] * inv;
}

// Build sampling locations (loc = ref + off/[W,H]) and broadcast aw to frames.
// i enumerates (f, q, m, l, p): i = ((((f*LQ+q)*8+m)*4+l)*4+p)
__global__ void build_loc_aw(const float* __restrict__ off, const float* __restrict__ ref,
                             const float* __restrict__ aw128, float* __restrict__ loc_out,
                             float* __restrict__ aw_out) {
  const int i = blockIdx.x * blockDim.x + threadIdx.x;
  const int total = NF * LQ * NHEAD * NLVL * NPTS;
  if (i >= total) return;
  const int p = i & 3;
  const int l = (i >> 2) & 3;
  const int m = (i >> 4) & 7;
  const int qf = i >> 7;            // f*LQ + q
  const int q = qf % LQ;
  const size_t ob = (size_t)qf * D_MODEL + m * 32 + l * 8 + p * 2;
  const float Wl = (float)cW[l], Hl = (float)cH[l];
  const float rx = ref[((size_t)qf * NLVL + l) * 2 + 0];
  const float ry = ref[((size_t)qf * NLVL + l) * 2 + 1];
  loc_out[(size_t)i * 2 + 0] = rx + off[ob + 0] / Wl;
  loc_out[(size_t)i * 2 + 1] = ry + off[ob + 1] / Hl;
  aw_out[i] = aw128[(size_t)q * 128 + m * 16 + l * 4 + p];
}

// Deformable bilinear sampling + attention-weighted sum.
// One thread per (f, q, m, d) output element.
__global__ void deform_sample(const float* __restrict__ value, const float* __restrict__ loc,
                              const float* __restrict__ aw, float* __restrict__ out) {
  const int i = blockIdx.x * blockDim.x + threadIdx.x;
  const int total = NF * LQ * D_MODEL;
  if (i >= total) return;
  const int d = i & 31;
  const int m = (i >> 5) & 7;
  const int qf = i >> 8;            // f*LQ + q
  const int f = qf / LQ;

  const float* __restrict__ vb = value + (size_t)f * LEN_IN * D_MODEL + m * 32 + d;
  float acc = 0.f;
#pragma unroll
  for (int l = 0; l < NLVL; ++l) {
    const int H = cH[l], W = cW[l];
    const float* __restrict__ vl = vb + (size_t)cS[l] * D_MODEL;
#pragma unroll
    for (int p = 0; p < NPTS; ++p) {
      const size_t li = (((size_t)qf * NHEAD + m) * NLVL + l) * NPTS + p;
      const float a  = aw[li];
      const float x = loc[li * 2 + 0] * W - 0.5f;
      const float y = loc[li * 2 + 1] * H - 0.5f;
      const float x0f = floorf(x), y0f = floorf(y);
      const int x0 = (int)x0f, y0 = (int)y0f;
      const float wx1 = x - x0f, wx0 = 1.f - wx1;
      const float wy1 = y - y0f, wy0 = 1.f - wy1;
      float sv = 0.f;
#pragma unroll
      for (int dy = 0; dy < 2; ++dy) {
        const int yi = y0 + dy;
        if (yi < 0 || yi > H - 1) continue;
        const float wy = dy ? wy1 : wy0;
#pragma unroll
        for (int dx = 0; dx < 2; ++dx) {
          const int xi = x0 + dx;
          if (xi < 0 || xi > W - 1) continue;
          const float wx = dx ? wx1 : wx0;
          sv += wy * wx * vl[(size_t)(yi * W + xi) * D_MODEL];
        }
      }
      acc += a * sv;
    }
  }
  out[(size_t)qf * D_MODEL + m * 32 + d] = acc;
}

// logits[f*LQ+q] = tb3[f,q,:] . tw_w + tw_b
__global__ void tw_logit(const float* __restrict__ tb3, const float* __restrict__ tww,
                         const float* __restrict__ twb, float* __restrict__ logits) {
  const int r = blockIdx.x * blockDim.x + threadIdx.x;
  if (r >= NF * LQ) return;
  const float* x = tb3 + (size_t)r * D_MODEL;
  float s = twb[0];
  for (int c = 0; c < D_MODEL; ++c) s += x[c] * tww[c];
  logits[r] = s;
}

// out[q,c] = sum_f t2[f,q,c] * softmax_f(logits[:,q])
__global__ void temporal_agg(const float* __restrict__ t2, const float* __restrict__ logits,
                             float* __restrict__ out) {
  const int i = blockIdx.x * blockDim.x + threadIdx.x;
  if (i >= LQ * D_MODEL) return;
  const int q = i >> 8;
  float lg[NF];
  float m = -3.4e38f;
#pragma unroll
  for (int f = 0; f < NF; ++f) { lg[f] = logits[f * LQ + q]; m = fmaxf(m, lg[f]); }
  float s = 0.f;
#pragma unroll
  for (int f = 0; f < NF; ++f) { lg[f] = __expf(lg[f] - m); s += lg[f]; }
  const float inv = 1.f / s;
  float acc = 0.f;
#pragma unroll
  for (int f = 0; f < NF; ++f) acc += t2[(size_t)f * LQ * D_MODEL + i] * lg[f] * inv;
  out[i] = acc;
}

// ---------------------------------------------------------------------------
// Host orchestration
// ---------------------------------------------------------------------------
extern "C" void kernel_launch(void* const* d_in, const int* in_sizes, int n_in,
                              void* d_out, int out_size, void* d_ws, size_t ws_size,
                              hipStream_t stream) {
  (void)in_sizes; (void)n_in; (void)out_size; (void)ws_size;

  const float* target    = (const float*)d_in[0];
  const float* tgt_box   = (const float*)d_in[1];   // (1,5,300,256) -> (1500,256)
  const float* query_pos = (const float*)d_in[2];
  const float* refpts    = (const float*)d_in[3];   // (1,5,300,4,2)
  const float* src       = (const float*)d_in[4];   // (5,20197,256)
  // d_in[5]=src_spatial_shapes, d_in[6]=level_start_index: static, baked in.
  const float* sa_in_w  = (const float*)d_in[7];
  const float* sa_in_b  = (const float*)d_in[8];
  const float* sa_out_w = (const float*)d_in[9];
  const float* sa_out_b = (const float*)d_in[10];
  const float* sab_in_w  = (const float*)d_in[11];
  const float* sab_in_b  = (const float*)d_in[12];
  const float* sab_out_w = (const float*)d_in[13];
  const float* sab_out_b = (const float*)d_in[14];
  const float* ln1_g  = (const float*)d_in[15];
  const float* ln1_b  = (const float*)d_in[16];
  const float* ln1b_g = (const float*)d_in[17];
  const float* ln1b_b = (const float*)d_in[18];
  const float* ln2_g  = (const float*)d_in[19];
  const float* ln2_b  = (const float*)d_in[20];
  const float* ln2b_g = (const float*)d_in[21];
  const float* ln2b_b = (const float*)d_in[22];
  const float* ln3_g  = (const float*)d_in[23];
  const float* ln3_b  = (const float*)d_in[24];
  const float* ln3b_g = (const float*)d_in[25];
  const float* ln3b_b = (const float*)d_in[26];
  const float* off_w  = (const float*)d_in[27];
  const float* off_b  = (const float*)d_in[28];
  const float* aw_w   = (const float*)d_in[29];
  const float* aw_b   = (const float*)d_in[30];
  const float* val_w  = (const float*)d_in[31];
  const float* val_b  = (const float*)d_in[32];
  const float* outp_w  = (const float*)d_in[33];
  const float* outp_b  = (const float*)d_in[34];
  const float* outpb_w = (const float*)d_in[35];
  const float* outpb_b = (const float*)d_in[36];
  const float* ffn1_w  = (const float*)d_in[37];
  const float* ffn1_b  = (const float*)d_in[38];
  const float* ffn1b_w = (const float*)d_in[39];
  const float* ffn1b_b = (const float*)d_in[40];
  const float* ffn2_w  = (const float*)d_in[41];
  const float* ffn2_b  = (const float*)d_in[42];
  const float* ffn2b_w = (const float*)d_in[43];
  const float* ffn2b_b = (const float*)d_in[44];
  const float* tw_w = (const float*)d_in[45];
  const float* tw_b = (const float*)d_in[46];

  float* out_target = (float*)d_out;                 //   300*256
  float* out_tb     = out_target + 76800;            //  1500*256
  float* out_loc    = out_tb + 384000;               //  192000*2
  float* out_aw     = out_loc + 384000;              //  192000

  // workspace cursor (float granularity; bf16 buffers take n/2 floats)
  float* ws = (float*)d_ws;
  size_t cur = 0;
  auto alloc  = [&](size_t n) { float* p = ws + cur; cur += n; return p; };
  auto allocb = [&](size_t n) { bf16_t* p = (bf16_t*)(ws + cur); cur += (n + 1) / 2; return p; };

  // fp32 intermediates
  float* q1      = alloc(76800);
  float* saq     = alloc(76800);
  float* sak     = alloc(76800);
  float* sav     = alloc(76800);
  float* sao     = alloc(76800);
  float* sat2    = alloc(76800);
  float* target1 = alloc(76800);
  float* query   = alloc(76800);
  float* qb   = alloc(384000);
  float* bq   = alloc(384000);
  float* bk   = alloc(384000);
  float* bvv  = alloc(384000);
  float* bo   = alloc(384000);
  float* bt2  = alloc(384000);
  float* tb1  = alloc(384000);
  float* qbox = alloc(384000);
  float* value = alloc((size_t)NF * LEN_IN * D_MODEL);   // 25.85M floats
  float* offb  = alloc(384000);
  float* awraw = alloc(38400);
  float* awsm  = alloc(38400);
  float* samp  = alloc(384000);
  float* t2c   = alloc(384000);
  float* tb2x  = alloc(384000);
  float* tb2   = alloc(384000);
  float* hb    = alloc((size_t)NF * LQ * D_FFN);         // 1500x1024
  float* ffnbo = alloc(384000);
  float* logits  = alloc(1500);
  float* t2agg   = alloc(76800);
  float* target2 = alloc(76800);
  float* h1      = alloc((size_t)LQ * D_FFN);            // 300x1024
  float* ffno    = alloc(76800);

  // bf16 weights (converted once per launch)
  bf16_t* w_sa_in   = allocb(768 * 256);
  bf16_t* w_sa_out  = allocb(256 * 256);
  bf16_t* w_sab_in  = allocb(768 * 256);
  bf16_t* w_sab_out = allocb(256 * 256);
  bf16_t* w_off   = allocb(256 * 256);
  bf16_t* w_aw    = allocb(128 * 256);
  bf16_t* w_val   = allocb(256 * 256);
  bf16_t* w_outp  = allocb(256 * 256);
  bf16_t* w_outpb = allocb(256 * 256);
  bf16_t* w_ffn1  = allocb(1024 * 256);
  bf16_t* w_ffn2  = allocb(256 * 1024);
  bf16_t* w_ffn1b = allocb(1024 * 256);
  bf16_t* w_ffn2b = allocb(256 * 1024);

  // bf16 activations
  bf16_t* b_target  = allocb(76800);
  bf16_t* b_q1      = allocb(76800);
  bf16_t* b_sao     = allocb(76800);
  bf16_t* b_query   = allocb(76800);
  bf16_t* b_tgt_box = allocb(384000);
  bf16_t* b_qb      = allocb(384000);
  bf16_t* b_bo      = allocb(384000);
  bf16_t* b_qbox    = allocb(384000);
  bf16_t* b_src     = allocb((size_t)NF * LEN_IN * D_MODEL);
  bf16_t* b_samp    = allocb(384000);
  bf16_t* b_tb2     = allocb(384000);
  bf16_t* b_hb      = allocb((size_t)NF * LQ * D_FFN);
  bf16_t* b_target2 = allocb(76800);
  bf16_t* b_h1      = allocb((size_t)LQ * D_FFN);

  auto cvt = [&](const float* in, bf16_t* out, int n) {
    cvt_bf16<<<(n / 8 + 255) / 256, 256, 0, stream>>>(in, out, n);
  };
  auto gemm = [&](const bf16_t* A, const bf16_t* W, const float* bias, float* C,
                  int M, int N, int K, int relu) {
    const int blkcnt = ((M + 15) / 16) * (N / 128);      // NT=8 -> 128-wide strips
    const int blocks = (blkcnt + 3) / 4;                 // 4 waves / block
    if (K == 256)
      gemm_bf16_wmma<8, 256><<<blocks, 128, 0, stream>>>(A, W, bias, C, M, N, relu);
    else
      gemm_bf16_wmma<8, 1024><<<blocks, 128, 0, stream>>>(A, W, bias, C, M, N, relu);
  };
  auto waves = [&](int w) { return ((size_t)w * 32 + 255) / 256; };

  // ---- weight conversion (once per launch) ----
  cvt(sa_in_w,   w_sa_in,   768 * 256);
  cvt(sa_out_w,  w_sa_out,  256 * 256);
  cvt(sab_in_w,  w_sab_in,  768 * 256);
  cvt(sab_out_w, w_sab_out, 256 * 256);
  cvt(off_w,   w_off,   256 * 256);
  cvt(aw_w,    w_aw,    128 * 256);
  cvt(val_w,   w_val,   256 * 256);
  cvt(outp_w,  w_outp,  256 * 256);
  cvt(outpb_w, w_outpb, 256 * 256);
  cvt(ffn1_w,  w_ffn1,  1024 * 256);
  cvt(ffn2_w,  w_ffn2,  256 * 1024);
  cvt(ffn1b_w, w_ffn1b, 1024 * 256);
  cvt(ffn2b_w, w_ffn2b, 256 * 1024);

  // ---- Stage A: instance self-attention (B=1, L=300) ----
  add_bc<<<(76800 + 255) / 256, 256, 0, stream>>>(target, query_pos, q1, 76800, 76800);
  cvt(q1, b_q1, 76800);
  cvt(target, b_target, 76800);
  gemm(b_q1,     w_sa_in,             sa_in_b,       saq, LQ, D_MODEL, D_MODEL, 0);
  gemm(b_q1,     w_sa_in + 256 * 256, sa_in_b + 256, sak, LQ, D_MODEL, D_MODEL, 0);
  gemm(b_target, w_sa_in + 512 * 256, sa_in_b + 512, sav, LQ, D_MODEL, D_MODEL, 0);
  mha_attn<<<waves(1 * NHEAD * LQ), 256, 0, stream>>>(saq, sak, sav, sao, 1, LQ, LQ);
  cvt(sao, b_sao, 76800);
  gemm(b_sao, w_sa_out, sa_out_b, sat2, LQ, D_MODEL, D_MODEL, 0);
  add_ln<<<waves(LQ), 256, 0, stream>>>(target, sat2, ln2_g, ln2_b, target1, LQ);

  // ---- Stage B: per-frame box self-attention (B=5, L=300) ----
  add_bc<<<(384000 + 255) / 256, 256, 0, stream>>>(tgt_box, query_pos, qb, 384000, 76800);
  cvt(qb, b_qb, 384000);
  cvt(tgt_box, b_tgt_box, 384000);
  gemm(b_qb,      w_sab_in,             sab_in_b,       bq,  NF * LQ, D_MODEL, D_MODEL, 0);
  gemm(b_qb,      w_sab_in + 256 * 256, sab_in_b + 256, bk,  NF * LQ, D_MODEL, D_MODEL, 0);
  gemm(b_tgt_box, w_sab_in + 512 * 256, sab_in_b + 512, bvv, NF * LQ, D_MODEL, D_MODEL, 0);
  mha_attn<<<waves(NF * NHEAD * LQ), 256, 0, stream>>>(bq, bk, bvv, bo, NF, LQ, LQ);
  cvt(bo, b_bo, 384000);
  gemm(b_bo, w_sab_out, sab_out_b, bt2, NF * LQ, D_MODEL, D_MODEL, 0);
  add_ln<<<waves(NF * LQ), 256, 0, stream>>>(tgt_box, bt2, ln2b_g, ln2b_b, tb1, NF * LQ);

  // ---- Stage C: deformable cross-attention ----
  add_bc<<<(76800 + 255) / 256, 256, 0, stream>>>(target1, query_pos, query, 76800, 76800);
  cvt(query, b_query, 76800);
  add_bc<<<(384000 + 255) / 256, 256, 0, stream>>>(tb1, query_pos, qbox, 384000, 76800);
  cvt(qbox, b_qbox, 384000);
  // value projection: the big GEMM (100985 x 256 x 256)
  cvt(src, b_src, NF * LEN_IN * D_MODEL);
  gemm(b_src, w_val, val_b, value, NF * LEN_IN, D_MODEL, D_MODEL, 0);
  gemm(b_qbox,  w_off, off_b, offb,  NF * LQ, D_MODEL, D_MODEL, 0);
  gemm(b_query, w_aw,  aw_b,  awraw, LQ, 128, D_MODEL, 0);
  softmax16<<<(LQ * NHEAD + 255) / 256, 256, 0, stream>>>(awraw, awsm, LQ * NHEAD);
  {
    const int total = NF * LQ * NHEAD * NLVL * NPTS;
    build_loc_aw<<<(total + 255) / 256, 256, 0, stream>>>(offb, refpts, awsm, out_loc, out_aw);
  }
  deform_sample<<<(NF * LQ * D_MODEL + 255) / 256, 256, 0, stream>>>(value, out_loc, out_aw, samp);
  cvt(samp, b_samp, 384000);
  gemm(b_samp, w_outp,  outp_b,  t2c,  NF * LQ, D_MODEL, D_MODEL, 0);
  gemm(b_samp, w_outpb, outpb_b, tb2x, NF * LQ, D_MODEL, D_MODEL, 0);

  // ---- Stage D: box-branch LN + FFN -> tb output ----
  add_ln<<<waves(NF * LQ), 256, 0, stream>>>(tb1, tb2x, ln1b_g, ln1b_b, tb2, NF * LQ);
  cvt(tb2, b_tb2, 384000);
  gemm(b_tb2, w_ffn1b, ffn1b_b, hb, NF * LQ, D_FFN, D_MODEL, 1);
  cvt(hb, b_hb, NF * LQ * D_FFN);
  gemm(b_hb, w_ffn2b, ffn2b_b, ffnbo, NF * LQ, D_MODEL, D_FFN, 0);
  add_ln<<<waves(NF * LQ), 256, 0, stream>>>(tb2, ffnbo, ln3b_g, ln3b_b, out_tb, NF * LQ);

  // ---- temporal aggregation + instance-branch LN + FFN -> target output ----
  tw_logit<<<(NF * LQ + 255) / 256, 256, 0, stream>>>(out_tb, tw_w, tw_b, logits);
  temporal_agg<<<(76800 + 255) / 256, 256, 0, stream>>>(t2c, logits, t2agg);
  add_ln<<<waves(LQ), 256, 0, stream>>>(target1, t2agg, ln1_g, ln1_b, target2, LQ);
  cvt(target2, b_target2, 76800);
  gemm(b_target2, w_ffn1, ffn1_b, h1, LQ, D_FFN, D_MODEL, 1);
  cvt(h1, b_h1, LQ * D_FFN);
  gemm(b_h1, w_ffn2, ffn2_b, ffno, LQ, D_MODEL, D_FFN, 0);
  add_ln<<<waves(LQ), 256, 0, stream>>>(target2, ffno, ln3_g, ln3_b, out_target, LQ);
}
